// GeneticDosageGNN_50405736186243
// MI455X (gfx1250) — compile-verified, matrix operand order checked by reading, not run on hardware
//
#include <hip/hip_runtime.h>

typedef __attribute__((ext_vector_type(16))) _Float16 v16h;
typedef __attribute__((ext_vector_type(8)))  float    v8f;

#define N_NODES  50000
#define N_EDGES  800000
#define N_GRAPHS 64
#define BN_EPS   1e-5f
#define ROW_TILES (N_NODES / 16)   // 50000 = 16 * 3125 exactly

// Guaranteed hardware float atomic (CDNA5 GLOBAL_ATOMIC_ADD_F32, no CAS loop).
// No return value needed -> non-returning atomic, tracked by STOREcnt; the
// implicit s_wait_idle at s_endpgm covers completion before the next kernel.
__device__ __forceinline__ void atomAddF(float* p, float v) {
    asm volatile("global_atomic_add_f32 %0, %1, off" : : "v"(p), "v"(v) : "memory");
}

// ---------------------------------------------------------------------------
// agg[dst[e], :] += x[src[e], :]   (one thread per (edge, float4-chunk))
// L2-resident scatter: 51 MB agg + 25-51 MB x fit the 192 MB L2.
// ---------------------------------------------------------------------------
__global__ void edge_scatter_add(const float* __restrict__ x,
                                 const int* __restrict__ src,
                                 const int* __restrict__ dst,
                                 float* __restrict__ agg,
                                 int fin) {
    const int cpe = fin >> 2;                     // float4 chunks per edge
    const long long total  = (long long)N_EDGES * cpe;
    const long long stride = (long long)gridDim.x * blockDim.x;
    for (long long t = (long long)blockIdx.x * blockDim.x + threadIdx.x;
         t < total; t += stride) {
        const int e = (int)(t / cpe);
        const int c = (int)(t % cpe);
        const int s = src[e];
        const int d = dst[e];
        const float4 val = *(const float4*)(x + (size_t)s * fin + 4 * c);
        float* ap = agg + (size_t)d * fin + 4 * c;
        atomAddF(ap + 0, val.x);
        atomAddF(ap + 1, val.y);
        atomAddF(ap + 2, val.z);
        atomAddF(ap + 3, val.w);
    }
}

// ---------------------------------------------------------------------------
// out = relu(bn(agg @ Wrel + brel + hin @ Wroot))
// Block = 256 threads = 8 waves; wave w -> 16-row tile; blockIdx.y -> 16 cols.
// B fragments (both weight matrices, all K-steps) staged to LDS once.
// ---------------------------------------------------------------------------
__global__ void gconv_wmma_bn_relu(const float* __restrict__ agg,
                                   const float* __restrict__ hin,
                                   const float* __restrict__ Wrel,
                                   const float* __restrict__ brel,
                                   const float* __restrict__ Wroot,
                                   const float* __restrict__ bng,
                                   const float* __restrict__ bnb,
                                   const float* __restrict__ bnm,
                                   const float* __restrict__ bnv,
                                   float* __restrict__ out,
                                   int fin, int fout) {
    __shared__ __attribute__((aligned(32))) _Float16 sB[2 * 8 * 32 * 16]; // 16 KB max

    const int ksteps = fin >> 5;                 // K in steps of 32
    const int lane   = threadIdx.x & 31;
    const int wave   = threadIdx.x >> 5;
    const int n      = lane & 15;
    const int hi     = lane >> 4;                // lane half selects K sub-range
    const int ncol   = blockIdx.y * 16 + n;

    // Stage B fragments into LDS in WMMA lane layout: half h -> K = idx + 8*hi + 16*grp
    for (int p = wave; p < 2 * ksteps; p += 8) {
        const int wsel = p & 1;                  // 0 = Wrel, 1 = Wroot
        const int ks   = p >> 1;
        const float* W = wsel ? Wroot : Wrel;
        _Float16* dp = sB + (((wsel * 8 + ks) * 32) + lane) * 16;
        const int kbase = ks * 32;
#pragma unroll
        for (int h = 0; h < 16; ++h) {
            const int grp = h >> 3, idx = h & 7;
            const int k = kbase + idx + 8 * hi + 16 * grp;
            dp[h] = (_Float16)W[(size_t)k * fout + ncol];
        }
    }
    __syncthreads();

    const int tile = blockIdx.x * 8 + wave;      // wave-uniform guard: EXEC stays all-ones
    if (tile >= ROW_TILES) return;
    const int row0 = tile * 16;
    const int m    = lane & 15;
    const float* rowA = agg + (size_t)(row0 + m) * fin;
    const float* rowH = hin + (size_t)(row0 + m) * fin;

    v8f c = {0.f, 0.f, 0.f, 0.f, 0.f, 0.f, 0.f, 0.f};
    for (int ks = 0; ks < ksteps; ++ks) {
        const int kbase = ks * 32;
        v16h a0, a1;
        // Halves 0-7 cover contiguous K = kbase+8*hi .. +7; halves 8-15 cover
        // kbase+16+8*hi .. +7  ->  4x b128 loads per fragment.
#pragma unroll
        for (int part = 0; part < 2; ++part) {
            const int kofs = kbase + 16 * part + 8 * hi;
            const float4 qa0 = *(const float4*)(rowA + kofs);
            const float4 qa1 = *(const float4*)(rowA + kofs + 4);
            const float4 qh0 = *(const float4*)(rowH + kofs);
            const float4 qh1 = *(const float4*)(rowH + kofs + 4);
            const int b = 8 * part;
            a0[b + 0] = (_Float16)qa0.x; a0[b + 1] = (_Float16)qa0.y;
            a0[b + 2] = (_Float16)qa0.z; a0[b + 3] = (_Float16)qa0.w;
            a0[b + 4] = (_Float16)qa1.x; a0[b + 5] = (_Float16)qa1.y;
            a0[b + 6] = (_Float16)qa1.z; a0[b + 7] = (_Float16)qa1.w;
            a1[b + 0] = (_Float16)qh0.x; a1[b + 1] = (_Float16)qh0.y;
            a1[b + 2] = (_Float16)qh0.z; a1[b + 3] = (_Float16)qh0.w;
            a1[b + 4] = (_Float16)qh1.x; a1[b + 5] = (_Float16)qh1.y;
            a1[b + 6] = (_Float16)qh1.z; a1[b + 7] = (_Float16)qh1.w;
        }
        const v16h b0 = *(const v16h*)(sB + (((0 * 8 + ks) * 32) + lane) * 16);
        const v16h b1 = *(const v16h*)(sB + (((1 * 8 + ks) * 32) + lane) * 16);
        c = __builtin_amdgcn_wmma_f32_16x16x32_f16(false, a0, false, b0, (short)0, c, false, false);
        c = __builtin_amdgcn_wmma_f32_16x16x32_f16(false, a1, false, b1, (short)0, c, false, false);
    }

    // Fused epilogue: bias + eval-BN folded to scale/shift + ReLU
    const float bias  = brel[ncol];
    const float scale = bng[ncol] * rsqrtf(bnv[ncol] + BN_EPS);
    const float shift = bnb[ncol] - bnm[ncol] * scale;
#pragma unroll
    for (int r = 0; r < 8; ++r) {
        const int row = row0 + r + 8 * hi;       // C/D layout: VGPR r -> M = r (+8 for hi lanes)
        float val = (c[r] + bias) * scale + shift;
        out[(size_t)row * fout + ncol] = val > 0.f ? val : 0.f;
    }
}

// ---------------------------------------------------------------------------
// pooled[batch[n], :] += h[n, :];  cnt[batch[n]] += 1
// ---------------------------------------------------------------------------
__global__ void pool_scatter(const float* __restrict__ h,
                             const int* __restrict__ batch,
                             float* __restrict__ pooled,
                             float* __restrict__ cnt) {
    const long long total  = (long long)N_NODES * 16;      // 16 float4 chunks / node
    const long long stride = (long long)gridDim.x * blockDim.x;
    for (long long t = (long long)blockIdx.x * blockDim.x + threadIdx.x;
         t < total; t += stride) {
        const int node  = (int)(t >> 4);
        const int chunk = (int)(t & 15);
        const int g = batch[node];
        const float4 v4 = *(const float4*)(h + (size_t)node * 64 + 4 * chunk);
        float* pp = pooled + g * 64 + 4 * chunk;
        atomAddF(pp + 0, v4.x);
        atomAddF(pp + 1, v4.y);
        atomAddF(pp + 2, v4.z);
        atomAddF(pp + 3, v4.w);
        if (chunk == 0) atomAddF(cnt + g, 1.0f);
    }
}

// ---------------------------------------------------------------------------
// out[g, c] = (pooled[g,:]/cnt[g]) @ W_out[:,c] + b_out[c]   (64x3 tiny head)
// ---------------------------------------------------------------------------
__global__ void final_head(const float* __restrict__ pooled,
                           const float* __restrict__ cnt,
                           const float* __restrict__ Wout,
                           const float* __restrict__ bout,
                           float* __restrict__ out) {
    const int t = threadIdx.x;
    if (t >= N_GRAPHS * 3) return;
    const int g = t / 3, cc = t % 3;
    const float inv = 1.0f / fmaxf(cnt[g], 1.0f);
    float acc = bout[cc];
    for (int f = 0; f < 64; ++f)
        acc += pooled[g * 64 + f] * inv * Wout[f * 3 + cc];
    out[t] = acc;
}

// ---------------------------------------------------------------------------
extern "C" void kernel_launch(void* const* d_in, const int* in_sizes, int n_in,
                              void* d_out, int out_size, void* d_ws, size_t ws_size,
                              hipStream_t stream) {
    const float* x     = (const float*)d_in[0];
    const int*   src   = (const int*)d_in[1];
    const int*   dst   = src + N_EDGES;
    const int*   batch = (const int*)d_in[2];

    const float *Wrel[3], *brel[3], *Wroot[3], *bng[3], *bnb[3], *bnm[3], *bnv[3];
    for (int i = 0; i < 3; ++i) {
        const int base = 3 + i * 7;
        Wrel[i]  = (const float*)d_in[base + 0];
        brel[i]  = (const float*)d_in[base + 1];
        Wroot[i] = (const float*)d_in[base + 2];
        bng[i]   = (const float*)d_in[base + 3];
        bnb[i]   = (const float*)d_in[base + 4];
        bnm[i]   = (const float*)d_in[base + 5];
        bnv[i]   = (const float*)d_in[base + 6];
    }
    const float* Wout = (const float*)d_in[24];
    const float* bout = (const float*)d_in[25];

    // Workspace layout (floats): agg[N*256] | h1[N*256] | h2[N*128] | h3[N*64] | pooled[64*64] | cnt[64]
    float* ws     = (float*)d_ws;
    float* agg    = ws;
    float* h1     = agg + (size_t)N_NODES * 256;
    float* h2     = h1  + (size_t)N_NODES * 256;
    float* h3     = h2  + (size_t)N_NODES * 128;
    float* pooled = h3  + (size_t)N_NODES * 64;
    float* cnt    = pooled + N_GRAPHS * 64;

    const int    fins[3]  = {128, 256, 128};
    const int    fouts[3] = {256, 128, 64};
    const float* ins[3]   = {x, h1, h2};
    float*       outs[3]  = {h1, h2, h3};

    for (int i = 0; i < 3; ++i) {
        const int fin = fins[i], fout = fouts[i];
        hipMemsetAsync(agg, 0, (size_t)N_NODES * fin * sizeof(float), stream);

        const long long work = (long long)N_EDGES * (fin / 4);
        const int sblocks = (int)((work + 255) / 256);
        edge_scatter_add<<<sblocks, 256, 0, stream>>>(ins[i], src, dst, agg, fin);

        dim3 grid((ROW_TILES + 7) / 8, fout / 16);
        gconv_wmma_bn_relu<<<grid, 256, 0, stream>>>(agg, ins[i],
                                                     Wrel[i], brel[i], Wroot[i],
                                                     bng[i], bnb[i], bnm[i], bnv[i],
                                                     outs[i], fin, fout);
    }

    hipMemsetAsync(pooled, 0, (size_t)(N_GRAPHS * 64 + N_GRAPHS) * sizeof(float), stream);
    const long long pwork = (long long)N_NODES * 16;
    pool_scatter<<<(int)((pwork + 255) / 256), 256, 0, stream>>>(h3, batch, pooled, cnt);
    final_head<<<1, 256, 0, stream>>>(pooled, cnt, Wout, bout, (float*)d_out);
}
